// TransGRUNet_2_4_89747636617849
// MI455X (gfx1250) — matrix operand order repaired and checked by reading, not compiled, not run on hardware
//
#include <hip/hip_runtime.h>
#include <hip/hip_bf16.h>
#include <math.h>

typedef unsigned short u16;
typedef unsigned int   u32;
typedef __attribute__((ext_vector_type(16))) __bf16 v16bf;
typedef __attribute__((ext_vector_type(8)))  float  v8f;

#define NN    128000
#define BB    128
#define LL    4
#define HH    128
#define EE    1024000
#define GG    (BB*LL)      // 512 groups
#define NPG   (NN/BB)      // 1000 nodes per graph

union Frag {
  v16bf v;
  u32   u[8];
  uint4 q[2];
};

__device__ __forceinline__ float wave_sum(float v) {
#pragma unroll
  for (int off = 16; off > 0; off >>= 1) v += __shfl_xor(v, off, 32);
  return v;
}

__device__ __forceinline__ u16 f32_to_bf16(float f) {
  u32 u = __float_as_uint(f);
  u32 r = (u + 0x7FFFu + ((u >> 16) & 1u)) >> 16;  // RNE
  return (u16)r;
}

// ---------------- utility kernels ----------------

__global__ void fill_u32(u32* p, u32 v, int n) {
  int i = blockIdx.x * blockDim.x + threadIdx.x;
  if (i < n) p[i] = v;
}

// Repack an fp32 weight matrix [K=128, Nc] (optionally stored transposed as
// [Nc,128]) into bf16 WMMA B-fragment order:
//   dst[((nt*4 + kc)*32 + lane)*8 + j] packs B[koff,col] | B[koff+1,col]<<16
// so each lane's fragment is 8 contiguous dwords (two b128 loads).
__global__ void repack_frag(const float* __restrict__ src, u32* __restrict__ dst,
                            int Nc, int transpose) {
  int idx = blockIdx.x * blockDim.x + threadIdx.x;
  int total = (Nc >> 4) * 4 * 32 * 8;
  if (idx >= total) return;
  int j    = idx & 7;
  int lane = (idx >> 3) & 31;
  int kc   = (idx >> 8) & 3;
  int nt   = idx >> 10;
  int half = lane >> 4;
  int l16  = lane & 15;
  int koff = 2 * (j & 3) + 16 * (j >> 2) + 8 * half + kc * 32;   // B 32x16: lanes 0-15 K 0-7,16-23; 16-31 K 8-15,24-31
  int col  = nt * 16 + l16;
  float f0 = transpose ? src[col * 128 + koff]     : src[koff * Nc + col];
  float f1 = transpose ? src[col * 128 + koff + 1] : src[(koff + 1) * Nc + col];
  dst[idx] = (u32)f32_to_bf16(f0) | ((u32)f32_to_bf16(f1) << 16);
}

__global__ void copy_cvt(const float* __restrict__ x, float* __restrict__ h,
                         u16* __restrict__ hbf, int n) {
  int i = blockIdx.x * blockDim.x + threadIdx.x;
  if (i >= n) return;
  float v = x[i];
  h[i] = v;
  hbf[i] = f32_to_bf16(v);
}

// ---------------- WMMA GEMM: C[M,Nc] = bf16(A[M,128]) @ Bpk + bias ----------
// A: row-major bf16 [M,128]; Bpk: fragment-packed (see repack_frag).
// Optional rowmap (gather A rows), optional posidx (adds sinusoidal pos-enc),
// outputs f32 (Cf) and/or bf16 (Cbf). One wave computes a 16-row stripe.

__global__ void gemm_wmma(const u16* __restrict__ A, const u32* __restrict__ Bpk,
                          const float* __restrict__ bias,
                          float* __restrict__ Cf, u16* __restrict__ Cbf,
                          int M, int Nc,
                          const int* __restrict__ rowmap,
                          const int* __restrict__ posidx) {
  int wave = threadIdx.x >> 5;
  int lane = threadIdx.x & 31;
  int half = lane >> 4;
  int l16  = lane & 15;
  int row0 = (blockIdx.x * 8 + wave) * 16;
  if (row0 >= M) return;

  int arow = row0 + l16;
  if (rowmap) arow = rowmap[arow];
  const u32* Abase = (const u32*)(A + (size_t)arow * 128);

  // A fragments for all 4 K-chunks (ISA 16-bit A 16x32 layout):
  // dword groups {0..3}+4h and {8..11}+4h within each 16-dword chunk.
  Frag afr[4];
#pragma unroll
  for (int kc = 0; kc < 4; ++kc) {
    afr[kc].q[0] = *(const uint4*)(Abase + kc * 16 + 4 * half);
    afr[kc].q[1] = *(const uint4*)(Abase + kc * 16 + 8 + 4 * half);
  }

  int ntiles = Nc >> 4;
  for (int nt = 0; nt < ntiles; ++nt) {
    v8f acc = {};
#pragma unroll
    for (int kc = 0; kc < 4; ++kc) {
      Frag bfr;
      const uint4* bp = (const uint4*)(Bpk + ((size_t)(nt * 4 + kc) * 32 + lane) * 8);
      bfr.q[0] = bp[0];
      bfr.q[1] = bp[1];
      acc = __builtin_amdgcn_wmma_f32_16x16x32_bf16(false, afr[kc].v, false, bfr.v,
                                                    (short)0, acc, false, false);
    }
    int col = nt * 16 + l16;
    float bv = bias[col];
    float pe_div = 0.f;
    if (posidx) pe_div = expf(-logf(10000.f) * (float)(col & ~1) * (1.f / 128.f));
#pragma unroll
    for (int r = 0; r < 8; ++r) {
      int row = row0 + r + 8 * half;
      float val = acc[r] + bv;
      if (posidx) {
        float ang = (float)posidx[row] * pe_div;
        val += (col & 1) ? cosf(ang) : sinf(ang);
      }
      if (Cf)  Cf[(size_t)row * Nc + col] = val;
      if (Cbf) Cbf[(size_t)row * Nc + col] = f32_to_bf16(val);
    }
  }
}

// ---------------- edge attention ----------------

__device__ __forceinline__ u32 enc_f32(float f) {
  u32 u = __float_as_uint(f);
  return (u & 0x80000000u) ? ~u : (u | 0x80000000u);
}
__device__ __forceinline__ float dec_f32(u32 e) {
  u32 u = (e & 0x80000000u) ? (e & 0x7FFFFFFFu) : ~e;
  return __uint_as_float(u);
}

__global__ void edge_logits(const float* __restrict__ q, const float* __restrict__ k,
                            const int* __restrict__ src, const int* __restrict__ dst,
                            float* __restrict__ logits, u32* __restrict__ segmax, int ne) {
  int e = (blockIdx.x * blockDim.x + threadIdx.x) >> 5;
  int lane = threadIdx.x & 31;
  if (e >= ne) return;
  int s = src[e], d = dst[e];
  const float4 qa = *(const float4*)(q + (size_t)d * HH + lane * 4);
  const float4 ka = *(const float4*)(k + (size_t)s * HH + lane * 4);
  float acc = qa.x * ka.x + qa.y * ka.y + qa.z * ka.z + qa.w * ka.w;
  acc = wave_sum(acc);
  if (lane == 0) {
    float lg = acc * 0.08838834764831845f;  // 1/sqrt(128)
    logits[e] = lg;
    atomicMax(segmax + d, enc_f32(lg));
  }
}

__global__ void edge_exp_sum(float* __restrict__ logits, const int* __restrict__ dst,
                             const u32* __restrict__ segmax, float* __restrict__ segsum,
                             int ne) {
  int e = blockIdx.x * blockDim.x + threadIdx.x;
  if (e >= ne) return;
  int d = dst[e];
  float w = expf(logits[e] - dec_f32(segmax[d]));
  logits[e] = w;
  atomicAdd(segsum + d, w);
}

__global__ void edge_agg(const float* __restrict__ ew, const float* __restrict__ segsum,
                         const float* __restrict__ v, const int* __restrict__ src,
                         const int* __restrict__ dst, float* __restrict__ agg, int ne) {
  int e = (blockIdx.x * blockDim.x + threadIdx.x) >> 5;
  int lane = threadIdx.x & 31;
  if (e >= ne) return;
  int s = src[e], d = dst[e];
  float alpha = ew[e] / segsum[d];
  const float4 va = *(const float4*)(v + (size_t)s * HH + lane * 4);
  float* aa = agg + (size_t)d * HH + lane * 4;
  atomicAdd(aa + 0, alpha * va.x);
  atomicAdd(aa + 1, alpha * va.y);
  atomicAdd(aa + 2, alpha * va.z);
  atomicAdd(aa + 3, alpha * va.w);
}

__global__ void ln_relu_cvt(const float* __restrict__ in, const float* __restrict__ g,
                            const float* __restrict__ b, float* __restrict__ h,
                            u16* __restrict__ hbf) {
  int n = (blockIdx.x * blockDim.x + threadIdx.x) >> 5;
  int lane = threadIdx.x & 31;
  if (n >= NN) return;
  const float* row = in + (size_t)n * HH + lane * 4;
  float x[4];
#pragma unroll
  for (int i = 0; i < 4; ++i) x[i] = row[i];
  float mu = wave_sum(x[0] + x[1] + x[2] + x[3]) * (1.f / HH);
  float s2 = 0.f;
#pragma unroll
  for (int i = 0; i < 4; ++i) { float d = x[i] - mu; s2 += d * d; }
  float var = wave_sum(s2) * (1.f / HH);
  float inv = rsqrtf(var + 1e-5f);
#pragma unroll
  for (int i = 0; i < 4; ++i) {
    int c = lane * 4 + i;
    float y = fmaxf((x[i] - mu) * inv * g[c] + b[c], 0.f);
    h[(size_t)n * HH + c] = y;
    hbf[(size_t)n * HH + c] = f32_to_bf16(y);
  }
}

// ---------------- clustering / grouping ----------------

__global__ void assign_levels(const float* __restrict__ h, const float* __restrict__ centers,
                              const int* __restrict__ batch, int* __restrict__ group,
                              int* __restrict__ counts) {
  int n = (blockIdx.x * blockDim.x + threadIdx.x) >> 5;
  int lane = threadIdx.x & 31;
  if (n >= NN) return;
  float hv[4];
#pragma unroll
  for (int i = 0; i < 4; ++i) hv[i] = h[(size_t)n * HH + lane * 4 + i];
  float best = -1e30f; int bl = 0;
#pragma unroll
  for (int c = 0; c < LL; ++c) {
    float d = 0.f, cn = 0.f;
#pragma unroll
    for (int i = 0; i < 4; ++i) {
      float cv = centers[c * HH + lane * 4 + i];
      d = fmaf(hv[i], cv, d);
      cn = fmaf(cv, cv, cn);
    }
    d = wave_sum(d); cn = wave_sum(cn);
    float sim = d / fmaxf(sqrtf(cn), 1e-8f);  // |h| scaling irrelevant for argmax
    if (sim > best) { best = sim; bl = c; }
  }
  if (lane == 0) {
    int g = batch[n] * LL + bl;
    group[n] = g;
    atomicAdd(counts + g, 1);
  }
}

__global__ void compute_pos(const int* __restrict__ group, const int* __restrict__ cseq,
                            int* __restrict__ pos) {
  int n = blockIdx.x * blockDim.x + threadIdx.x;
  if (n >= NN) return;
  int g = group[n], s0 = cseq[n];
  int base = (n / NPG) * NPG;
  int cnt = 0;
  for (int m = base; m < base + NPG; ++m)
    if (group[m] == g && cseq[m] < s0) cnt++;
  pos[n] = cnt;
}

__global__ void compute_starts(const int* __restrict__ counts, int* __restrict__ starts) {
  if (threadIdx.x == 0 && blockIdx.x == 0) {
    int run = 0;
    for (int g = 0; g < GG; ++g) { starts[g] = run; run += counts[g]; }
  }
}

__global__ void scatter_slots(const int* __restrict__ group, const int* __restrict__ pos,
                              const int* __restrict__ starts, int* __restrict__ slotnode,
                              int* __restrict__ posslot) {
  int n = blockIdx.x * blockDim.x + threadIdx.x;
  if (n >= NN) return;
  int slot = starts[group[n]] + pos[n];
  slotnode[slot] = n;
  posslot[slot] = pos[n];
}

// ---------------- GRU recurrence (one workgroup per group, causal truncation) ----

__global__ void gru_seq(const float* __restrict__ gi, const float* __restrict__ Whh,
                        const float* __restrict__ bhh, const int* __restrict__ counts,
                        const int* __restrict__ starts, float* __restrict__ outs) {
  int g = blockIdx.x;
  int c = threadIdx.x;  // 128 threads
  __shared__ float hsh[HH];
  hsh[c] = 0.f;
  __syncthreads();
  int len = counts[g];
  size_t base = (size_t)starts[g];
  const float* Wr = Whh + (size_t)c * HH;
  const float* Wz = Whh + (size_t)(HH + c) * HH;
  const float* Wn = Whh + (size_t)(2 * HH + c) * HH;
  float br = bhh[c], bz = bhh[HH + c], bn = bhh[2 * HH + c];
  for (int t = 0; t < len; ++t) {
    const float* girow = gi + (base + t) * (3 * HH);
    if (t + 1 < len) {  // hide next step's global latency (global_prefetch_b8)
      __builtin_prefetch(girow + 3 * HH + c, 0, 1);
    }
    float ghr = br, ghz = bz, ghn = bn;
#pragma unroll 8
    for (int kk = 0; kk < HH; ++kk) {
      float hv = hsh[kk];
      ghr = fmaf(hv, Wr[kk], ghr);
      ghz = fmaf(hv, Wz[kk], ghz);
      ghn = fmaf(hv, Wn[kk], ghn);
    }
    float r  = 1.f / (1.f + expf(-(girow[c] + ghr)));
    float z  = 1.f / (1.f + expf(-(girow[HH + c] + ghz)));
    float nn = tanhf(girow[2 * HH + c] + r * ghn);
    float hn = (1.f - z) * nn + z * hsh[c];
    __syncthreads();
    hsh[c] = hn;
    __syncthreads();
    outs[(base + t) * HH + c] = hn;
  }
}

// ---------------- head: LN+ReLU (non-singleton), gelu-MLP-sigmoid ----------------

__global__ void final_head(const float* __restrict__ h, const float* __restrict__ outs,
                           const int* __restrict__ group, const int* __restrict__ pos,
                           const int* __restrict__ counts, const int* __restrict__ starts,
                           const float* __restrict__ sg, const float* __restrict__ sb,
                           const float* __restrict__ W1, const float* __restrict__ b1,
                           const float* __restrict__ W2, const float* __restrict__ b2,
                           float* __restrict__ y) {
  int n = (blockIdx.x * blockDim.x + threadIdx.x) >> 5;
  int lane = threadIdx.x & 31;
  if (n >= NN) return;
  int g = group[n];
  bool single = (counts[g] == 1);
  float v[4];
  if (single) {
#pragma unroll
    for (int i = 0; i < 4; ++i) v[i] = h[(size_t)n * HH + lane * 4 + i];
  } else {
    size_t slot = (size_t)starts[g] + pos[n];
    float x[4];
#pragma unroll
    for (int i = 0; i < 4; ++i) x[i] = outs[slot * HH + lane * 4 + i];
    float mu = wave_sum(x[0] + x[1] + x[2] + x[3]) * (1.f / HH);
    float s2 = 0.f;
#pragma unroll
    for (int i = 0; i < 4; ++i) { float d = x[i] - mu; s2 += d * d; }
    float inv = rsqrtf(wave_sum(s2) * (1.f / HH) + 1e-5f);
#pragma unroll
    for (int i = 0; i < 4; ++i) {
      int c = lane * 4 + i;
      v[i] = fmaxf((x[i] - mu) * inv * sg[c] + sb[c], 0.f);
    }
  }
  float acc[8] = {0, 0, 0, 0, 0, 0, 0, 0};
#pragma unroll
  for (int i = 0; i < 4; ++i) {
    int c = lane * 4 + i;
#pragma unroll
    for (int j = 0; j < 8; ++j) acc[j] = fmaf(v[i], W1[c * 8 + j], acc[j]);
  }
#pragma unroll
  for (int j = 0; j < 8; ++j) acc[j] = wave_sum(acc[j]);
  float yy = b2[0];
#pragma unroll
  for (int j = 0; j < 8; ++j) {
    float a = acc[j] + b1[j];
    float ge = 0.5f * a * (1.f + erff(a * 0.70710678118654752f));  // exact gelu
    yy = fmaf(ge, W2[j], yy);
  }
  yy = 1.f / (1.f + expf(-yy));
  if (lane == 0) y[n] = yy;
}

__global__ void distance_k(const float* __restrict__ centers, float* __restrict__ out) {
  if (threadIdx.x || blockIdx.x) return;
  float d[6]; int t = 0;
  for (int i = 0; i < LL; ++i)
    for (int j = i + 1; j < LL; ++j) {
      float s = 0.f;
      for (int k = 0; k < HH; ++k) {
        float df = centers[i * HH + k] - centers[j * HH + k];
        s += df * df;
      }
      d[t++] = sqrtf(s);
    }
  float mu = 0.f;
  for (int i = 0; i < 6; ++i) mu += d[i];
  mu *= (1.f / 6.f);
  float var = 0.f;
  for (int i = 0; i < 6; ++i) { float e = d[i] - mu; var += e * e; }
  var *= (1.f / 5.f);  // unbiased
  out[0] = -var;
}

// ---------------- driver ----------------

extern "C" void kernel_launch(void* const* d_in, const int* in_sizes, int n_in,
                              void* d_out, int out_size, void* d_ws, size_t ws_size,
                              hipStream_t stream) {
  (void)in_sizes; (void)n_in; (void)out_size; (void)ws_size;
  const float* x    = (const float*)d_in[0];
  const int* tei    = (const int*)d_in[1];   // [2,E]: src=tei, dst=tei+E
  const int* cseq   = (const int*)d_in[2];
  const int* batch  = (const int*)d_in[3];
  const float* Wq   = (const float*)d_in[4];
  const float* bq   = (const float*)d_in[5];
  const float* Wk   = (const float*)d_in[6];
  const float* bk   = (const float*)d_in[7];
  const float* Wv   = (const float*)d_in[8];
  const float* bv   = (const float*)d_in[9];
  const float* Wsm  = (const float*)d_in[10];
  const float* bsm  = (const float*)d_in[11];
  const float* lng  = (const float*)d_in[12];
  const float* lnb  = (const float*)d_in[13];
  const float* cent = (const float*)d_in[14];
  const float* projW = (const float*)d_in[15];
  const float* projb = (const float*)d_in[16];
  const float* Wih  = (const float*)d_in[17];
  const float* Whh  = (const float*)d_in[18];
  const float* bih  = (const float*)d_in[19];
  const float* bhh  = (const float*)d_in[20];
  const float* sg   = (const float*)d_in[21];
  const float* sb   = (const float*)d_in[22];
  const float* W1   = (const float*)d_in[23];
  const float* b1   = (const float*)d_in[24];
  const float* W2   = (const float*)d_in[25];
  const float* b2   = (const float*)d_in[26];
  float* out = (float*)d_out;

  // workspace carve-out (deterministic layout each call)
  char* base = (char*)d_ws;
  size_t off = 0;
  auto alloc = [&](size_t bytes) -> char* {
    char* p = base + off; off += (bytes + 255) & ~(size_t)255; return p;
  };
  float* hA    = (float*)alloc((size_t)NN * HH * 4);
  u16*   hbf   = (u16*)  alloc((size_t)NN * HH * 2);
  float* qkv   = (float*)alloc((size_t)3 * NN * HH * 4);  // q,k,v; later reused as gi [N,384]
  float* qB = qkv;
  float* kB = qkv + (size_t)NN * HH;
  float* vB = qkv + (size_t)2 * NN * HH;
  float* gi = qkv;
  float* agg = (float*)alloc((size_t)NN * HH * 4);        // later reused as sin_bf
  u16* sinbf = (u16*)agg;
  float* logits = (float*)alloc((size_t)EE * 4);
  u32*   segmax = (u32*)alloc((size_t)NN * 4);
  float* segsum = (float*)alloc((size_t)NN * 4);
  int* group    = (int*)alloc((size_t)NN * 4);
  int* pos      = (int*)alloc((size_t)NN * 4);
  int* slotnode = (int*)alloc((size_t)NN * 4);
  int* posslot  = (int*)alloc((size_t)NN * 4);
  int* counts   = (int*)alloc(GG * 4);
  int* startsA  = (int*)alloc(GG * 4);
  float* outsC  = (float*)alloc((size_t)NN * HH * 4);
  u32* wpk      = (u32*)alloc((size_t)(13 * HH * HH + HH * 3 * HH) * 2);  // packed bf16

  dim3 blk(256);
  const int MMU = HH * HH / 2;       // 8192 dwords per packed 128x128 matrix
  const int WIHU = HH * 3 * HH / 2;  // 24576 dwords for packed 128x384

  // repack weights fp32 -> bf16 WMMA-fragment order
  for (int i = 0; i < 3; ++i) {
    repack_frag<<<(MMU + 255) / 256, blk, 0, stream>>>(Wq + (size_t)i * HH * HH, wpk + (size_t)(i * 4 + 0) * MMU, HH, 0);
    repack_frag<<<(MMU + 255) / 256, blk, 0, stream>>>(Wk + (size_t)i * HH * HH, wpk + (size_t)(i * 4 + 1) * MMU, HH, 0);
    repack_frag<<<(MMU + 255) / 256, blk, 0, stream>>>(Wv + (size_t)i * HH * HH, wpk + (size_t)(i * 4 + 2) * MMU, HH, 0);
    repack_frag<<<(MMU + 255) / 256, blk, 0, stream>>>(Wsm + (size_t)i * HH * HH, wpk + (size_t)(i * 4 + 3) * MMU, HH, 0);
  }
  repack_frag<<<(MMU + 255) / 256, blk, 0, stream>>>(projW, wpk + (size_t)12 * MMU, HH, 0);
  repack_frag<<<(WIHU + 255) / 256, blk, 0, stream>>>(Wih, wpk + (size_t)13 * MMU, 3 * HH, 1);  // Wih.T

  copy_cvt<<<(NN * HH + 255) / 256, blk, 0, stream>>>(x, hA, hbf, NN * HH);

  const int* esrc = tei;
  const int* edst = tei + EE;
  int gemm_grid = NN / 128;        // 8 waves/block x 16 rows/wave
  int wave_grid_N = NN * 32 / 256; // one wave per node
  int wave_grid_E = EE * 32 / 256; // one wave per edge

  for (int i = 0; i < 3; ++i) {
    const u32* wq = wpk + (size_t)(i * 4 + 0) * MMU;
    const u32* wk = wpk + (size_t)(i * 4 + 1) * MMU;
    const u32* wv = wpk + (size_t)(i * 4 + 2) * MMU;
    const u32* ws = wpk + (size_t)(i * 4 + 3) * MMU;
    gemm_wmma<<<gemm_grid, blk, 0, stream>>>(hbf, wq, bq + i * HH, qB, nullptr, NN, HH, nullptr, nullptr);
    gemm_wmma<<<gemm_grid, blk, 0, stream>>>(hbf, wk, bk + i * HH, kB, nullptr, NN, HH, nullptr, nullptr);
    gemm_wmma<<<gemm_grid, blk, 0, stream>>>(hbf, wv, bv + i * HH, vB, nullptr, NN, HH, nullptr, nullptr);
    // agg initialized with the root/skip term x@Ws+bs; edges atomically add on top
    gemm_wmma<<<gemm_grid, blk, 0, stream>>>(hbf, ws, bsm + i * HH, agg, nullptr, NN, HH, nullptr, nullptr);
    fill_u32<<<(NN + 255) / 256, blk, 0, stream>>>(segmax, 0u, NN);
    fill_u32<<<(NN + 255) / 256, blk, 0, stream>>>((u32*)segsum, 0u, NN);
    edge_logits<<<wave_grid_E, blk, 0, stream>>>(qB, kB, esrc, edst, logits, segmax, EE);
    edge_exp_sum<<<(EE + 255) / 256, blk, 0, stream>>>(logits, edst, segmax, segsum, EE);
    edge_agg<<<wave_grid_E, blk, 0, stream>>>(logits, segsum, vB, esrc, edst, agg, EE);
    ln_relu_cvt<<<wave_grid_N, blk, 0, stream>>>(agg, lng + i * HH, lnb + i * HH, hA, hbf);
  }

  // cluster assignment + compact slot layout
  fill_u32<<<(GG + 255) / 256, blk, 0, stream>>>((u32*)counts, 0u, GG);
  assign_levels<<<wave_grid_N, blk, 0, stream>>>(hA, cent, batch, group, counts);
  compute_pos<<<(NN + 255) / 256, blk, 0, stream>>>(group, cseq, pos);
  compute_starts<<<1, 1, 0, stream>>>(counts, startsA);
  scatter_slots<<<(NN + 255) / 256, blk, 0, stream>>>(group, pos, startsA, slotnode, posslot);

  // sin = h[slot] @ projW + projb + posenc  (gathered rows, bf16 out, into agg region)
  gemm_wmma<<<gemm_grid, blk, 0, stream>>>(hbf, wpk + (size_t)12 * MMU, projb,
                                           nullptr, sinbf, NN, HH, slotnode, posslot);
  // gi = sin @ Wih.T + bih  (f32 out, into qkv region)
  gemm_wmma<<<gemm_grid, blk, 0, stream>>>(sinbf, wpk + (size_t)13 * MMU, bih,
                                           gi, nullptr, NN, 3 * HH, nullptr, nullptr);
  gru_seq<<<GG, HH, 0, stream>>>(gi, Whh, bhh, counts, startsA, outsC);

  final_head<<<wave_grid_N, blk, 0, stream>>>(hA, outsC, group, pos, counts, startsA,
                                              sg, sb, W1, b1, W2, b2, out);
  distance_k<<<1, 1, 0, stream>>>(cent, out + NN);
}
